// GraphIsomorphismNetwork_13932873908319
// MI455X (gfx1250) — matrix operand order repaired and checked by reading, not compiled, not run on hardware
//
#include <hip/hip_runtime.h>
#include <hip/hip_bf16.h>
#include <stdint.h>

// ---------------- problem constants (match reference) ----------------
#define NN   100000
#define EE   400000
#define GG   2000
#define FIN  64
#define HH   128
#define NB8  8          // HH/16 output col-blocks
#define BN_EPS 1e-5f

// ---------------- WMMA types ----------------
typedef __attribute__((ext_vector_type(16))) __bf16 bf16x16;
typedef __attribute__((ext_vector_type(8)))  float  f32x8;

union FragAB { bf16x16 v; unsigned u[8]; uint4 q[2]; };

__device__ __forceinline__ unsigned short f32_to_bf16_rn(float f) {
  unsigned u = __float_as_uint(f);
  u += 0x7FFFu + ((u >> 16) & 1u);      // round-to-nearest-even
  return (unsigned short)(u >> 16);
}

// =====================================================================
// Weight pack: fp32 W[K x 128] (row-major, K = reduction dim) ->
// swizzled bf16 B-fragments, laid out so a wave's 16x16x32 WMMA B operand
// is 8 consecutive dwords per lane:
//   out[((kb*8 + nb)*32 + lane)*8 + r] = pack(W[k0, col], W[k0+1, col])
//   col = nb*16 + (lane&15),  k0 = kb*32 + (lane>>4)*16 + 2r
// (k order used here, same as in A: lanes 0-15 hold K 0-15, lanes 16-31 hold K 16-31)
// =====================================================================
__global__ void pack_w_kernel(const float* __restrict__ w,
                              unsigned* __restrict__ out, int K) {
  int tid   = blockIdx.x * blockDim.x + threadIdx.x;
  int total = (K >> 5) * NB8 * 32 * 8;
  if (tid >= total) return;
  int r    = tid & 7;
  int lane = (tid >> 3) & 31;
  int nb   = (tid >> 8) & 7;
  int kb   = tid >> 11;
  int col  = nb * 16 + (lane & 15);
  int k0   = kb * 32 + ((lane >> 4) << 4) + 2 * r;
  unsigned lo = f32_to_bf16_rn(w[(size_t)k0 * HH + col]);
  unsigned hi = f32_to_bf16_rn(w[(size_t)(k0 + 1) * HH + col]);
  out[tid] = lo | (hi << 16);
}

// =====================================================================
// Core WMMA helper: accumulate a 16x16 f32 tile of  A(16xKin) @ W(Kinx128)
// A tile lives in LDS as bf16, row-major with stride strideA (elements).
// A-fragment layout (16-bit A 16x32): lane half h = lane>>4, m = lane&15,
//   vgpr r holds K = kb*32 + 16h + 2r .. +1  (even k ->
//   one aligned 4-byte LDS load per vgpr).
// =====================================================================
__device__ __forceinline__ f32x8 wmma_rowtile(const unsigned short* sA,
                                              int strideA, int Kin,
                                              const unsigned* __restrict__ wsw,
                                              int nb, int lane, f32x8 acc) {
  const int half = lane >> 4;
  const int m    = lane & 15;
  for (int kb = 0; kb < (Kin >> 5); ++kb) {
    FragAB a, b;
#pragma unroll
    for (int r = 0; r < 8; ++r) {
      int k0 = kb * 32 + half * 16 + 2 * r;
      a.u[r] = *(const unsigned*)(sA + m * strideA + k0);
    }
    const uint4* bp =
        (const uint4*)(wsw + (((size_t)(kb * NB8 + nb) * 32 + lane) << 3));
    b.q[0] = bp[0];
    b.q[1] = bp[1];
    acc = __builtin_amdgcn_wmma_f32_16x16x32_bf16(
        false, a.v, false, b.v, (short)0, acc, false, false);
  }
  return acc;
}

// =====================================================================
// Fused GIN MLP:  out = [relu?] ( relu(z @ W1 + b1) @ W2 + b2 )
// One block (8 wave32) per 16-row node tile; wave nb -> output cols
// [16nb,16nb+16). z tile and intermediate tile staged in LDS as bf16.
// =====================================================================
__global__ void __launch_bounds__(256)
gin_mlp_kernel(const float* __restrict__ z, int Kin,
               const unsigned* __restrict__ w1sw, const float* __restrict__ b1,
               const unsigned* __restrict__ w2sw, const float* __restrict__ b2,
               float* __restrict__ out, int relu_out) {
  __shared__ unsigned short sA[16 * HH];
  __shared__ unsigned short sT[16 * HH];
  const int tid     = threadIdx.x;
  const int lane    = tid & 31;
  const int nb      = tid >> 5;
  const int rowbase = blockIdx.x * 16;

  // stage input tile -> LDS (bf16)
  for (int i = tid; i < 16 * Kin; i += 256) {
    int row = i / Kin, c = i - row * Kin;
    sA[row * Kin + c] = f32_to_bf16_rn(z[(size_t)(rowbase + row) * Kin + c]);
  }
  __syncthreads();

  const int half = lane >> 4;
  const int col  = nb * 16 + (lane & 15);

  f32x8 acc;
#pragma unroll
  for (int r = 0; r < 8; ++r) acc[r] = b1[col];
  acc = wmma_rowtile(sA, Kin, Kin, w1sw, nb, lane, acc);

  // relu + stage intermediate tile (C/D layout: vgpr r -> M = r + 8*half)
#pragma unroll
  for (int r = 0; r < 8; ++r) {
    float v = fmaxf(acc[r], 0.0f);
    sT[(r + half * 8) * HH + col] = f32_to_bf16_rn(v);
  }
  __syncthreads();

  f32x8 acc2;
#pragma unroll
  for (int r = 0; r < 8; ++r) acc2[r] = b2[col];
  acc2 = wmma_rowtile(sT, HH, HH, w2sw, nb, lane, acc2);

#pragma unroll
  for (int r = 0; r < 8; ++r) {
    float v = acc2[r];
    if (relu_out) v = fmaxf(v, 0.0f);
    out[(size_t)(rowbase + r + half * 8) * HH + col] = v;
  }
}

// ================= single-matmul WMMA GEMM: out = relu(in @ W + b) =====
__global__ void __launch_bounds__(256)
gemm_relu_kernel(const float* __restrict__ in, const unsigned* __restrict__ wsw,
                 const float* __restrict__ bias, float* __restrict__ out) {
  __shared__ unsigned short sA[16 * HH];
  const int tid     = threadIdx.x;
  const int lane    = tid & 31;
  const int nb      = tid >> 5;
  const int rowbase = blockIdx.x * 16;

  for (int i = tid; i < 16 * HH; i += 256) {
    int row = i >> 7, c = i & 127;
    sA[i] = f32_to_bf16_rn(in[(size_t)(rowbase + row) * HH + c]);
  }
  __syncthreads();

  const int half = lane >> 4;
  const int col  = nb * 16 + (lane & 15);
  f32x8 acc;
#pragma unroll
  for (int r = 0; r < 8; ++r) acc[r] = bias[col];
  acc = wmma_rowtile(sA, HH, HH, wsw, nb, lane, acc);
#pragma unroll
  for (int r = 0; r < 8; ++r)
    out[(size_t)(rowbase + r + half * 8) * HH + col] = fmaxf(acc[r], 0.0f);
}

// ---------------- elementwise / scatter kernels ----------------
__global__ void copy4_kernel(float* __restrict__ dst,
                             const float* __restrict__ src, size_t n4) {
  size_t i = (size_t)blockIdx.x * blockDim.x + threadIdx.x;
  if (i < n4) ((float4*)dst)[i] = ((const float4*)src)[i];
}

__global__ void zero_kernel(float* __restrict__ p, size_t n) {
  size_t i = (size_t)blockIdx.x * blockDim.x + threadIdx.x;
  if (i < n) p[i] = 0.0f;
}

// z[dst] += h[src] over all edges; z pre-initialized to h (GIN eps=0)
__global__ void edge_agg_kernel(const float* __restrict__ h,
                                const int* __restrict__ src,
                                const int* __restrict__ dst,
                                float* __restrict__ z, int F) {
  int t      = blockIdx.x * blockDim.x + threadIdx.x;
  int chunks = F >> 2;
  int e      = t / chunks;
  if (e >= EE) return;
  int c = (t - e * chunks) << 2;
  int s = src[e], d = dst[e];
  const float4 v = *(const float4*)(h + (size_t)s * F + c);
  float* p = z + (size_t)d * F + c;
  atomicAdd(p + 0, v.x);
  atomicAdd(p + 1, v.y);
  atomicAdd(p + 2, v.z);
  atomicAdd(p + 3, v.w);
}

__global__ void bn_relu_kernel(float* __restrict__ h, const float* __restrict__ g,
                               const float* __restrict__ b, const float* __restrict__ m,
                               const float* __restrict__ v, size_t total) {
  size_t i = (size_t)blockIdx.x * blockDim.x + threadIdx.x;
  if (i >= total) return;
  int f = (int)(i & (HH - 1));
  float val = (h[i] - m[f]) * rsqrtf(v[f] + BN_EPS) * g[f] + b[f];
  h[i] = fmaxf(val, 0.0f);
}

__global__ void pool_acc_kernel(const float* __restrict__ h,
                                const int* __restrict__ batch,
                                float* __restrict__ sums, float* __restrict__ cnt) {
  int t    = blockIdx.x * blockDim.x + threadIdx.x;
  int node = t >> 5;
  if (node >= NN) return;
  int c = (t & 31) << 2;
  int g = batch[node];
  const float4 v4 = *(const float4*)(h + (size_t)node * HH + c);
  float* p = sums + (size_t)g * HH + c;
  atomicAdd(p + 0, v4.x);
  atomicAdd(p + 1, v4.y);
  atomicAdd(p + 2, v4.z);
  atomicAdd(p + 3, v4.w);
  if (c == 0) atomicAdd(cnt + g, 1.0f);
}

__global__ void pool_div_kernel(float* __restrict__ sums,
                                const float* __restrict__ cnt, int total) {
  int i = blockIdx.x * blockDim.x + threadIdx.x;
  if (i < total) sums[i] /= fmaxf(cnt[i >> 7], 1.0f);
}

__global__ void head_kernel(const float* __restrict__ t,
                            const float* __restrict__ ow2,
                            const float* __restrict__ ob2,
                            float* __restrict__ out) {
  int g = blockIdx.x * blockDim.x + threadIdx.x;
  if (g >= GG) return;
  float acc = ob2[0];
#pragma unroll 4
  for (int f = 0; f < HH; ++f) acc += t[(size_t)g * HH + f] * ow2[f];
  out[g] = acc;
}

// =====================================================================
// host launcher
// =====================================================================
extern "C" void kernel_launch(void* const* d_in, const int* in_sizes, int n_in,
                              void* d_out, int out_size, void* d_ws, size_t ws_size,
                              hipStream_t stream) {
  (void)in_sizes; (void)n_in; (void)out_size; (void)ws_size;

  const float* x     = (const float*)d_in[0];
  const int*   ei    = (const int*)d_in[1];
  const int*   batch = (const int*)d_in[2];
  const int*   src   = ei;
  const int*   dstp  = ei + EE;

  // params flattened in dict insertion order
  const float* P[36];
  for (int i = 0; i < 36; ++i) P[i] = (const float*)d_in[3 + i];
  const float *c1w1 = P[0],  *c1b1 = P[1],  *c1w2 = P[2],  *c1b2 = P[3];
  const float *bn1g = P[4],  *bn1b = P[5],  *bn1m = P[6],  *bn1v = P[7];
  const float *c2aw1 = P[8], *c2ab1 = P[9], *c2aw2 = P[10], *c2ab2 = P[11];
  const float *c2bw1 = P[12],*c2bb1 = P[13],*c2bw2 = P[14], *c2bb2 = P[15];
  const float *bn2g = P[16], *bn2b = P[17], *bn2m = P[18],  *bn2v = P[19];
  const float *c3aw1 = P[20],*c3ab1 = P[21],*c3aw2 = P[22], *c3ab2 = P[23];
  const float *c3bw1 = P[24],*c3bb1 = P[25],*c3bw2 = P[26], *c3bb2 = P[27];
  const float *bn3g = P[28], *bn3b = P[29], *bn3m = P[30],  *bn3v = P[31];
  const float *ow1 = P[32],  *ob1 = P[33],  *ow2 = P[34],   *ob2 = P[35];

  // ---- workspace carve-up ----
  char* ws = (char*)d_ws;
  size_t off = 0;
  auto carve = [&](size_t bytes) -> void* {
    void* p = ws + off;
    off = (off + bytes + 255) & ~(size_t)255;
    return p;
  };
  float* hA    = (float*)carve((size_t)NN * HH * 4);
  float* hB    = (float*)carve((size_t)NN * HH * 4);
  float* zb    = (float*)carve((size_t)NN * HH * 4);
  float* sums  = (float*)carve((size_t)GG * HH * 4);
  float* cnt   = (float*)carve((size_t)GG * 4);
  float* tpool = (float*)carve((size_t)GG * HH * 4);
  unsigned* sw_c1w1  = (unsigned*)carve((size_t)FIN * HH * 2);
  unsigned* sw_c1w2  = (unsigned*)carve((size_t)HH * HH * 2);
  unsigned* sw_c2aw1 = (unsigned*)carve((size_t)HH * HH * 2);
  unsigned* sw_c2aw2 = (unsigned*)carve((size_t)HH * HH * 2);
  unsigned* sw_c2bw1 = (unsigned*)carve((size_t)HH * HH * 2);
  unsigned* sw_c2bw2 = (unsigned*)carve((size_t)HH * HH * 2);
  unsigned* sw_c3aw1 = (unsigned*)carve((size_t)HH * HH * 2);
  unsigned* sw_c3aw2 = (unsigned*)carve((size_t)HH * HH * 2);
  unsigned* sw_c3bw1 = (unsigned*)carve((size_t)HH * HH * 2);
  unsigned* sw_c3bw2 = (unsigned*)carve((size_t)HH * HH * 2);
  unsigned* sw_ow1   = (unsigned*)carve((size_t)HH * HH * 2);

  // ---- 1) pack weights to swizzled bf16 fragments ----
  auto pack = [&](const float* w, unsigned* o, int K) {
    int total = (K / 32) * NB8 * 32 * 8;
    pack_w_kernel<<<(total + 255) / 256, 256, 0, stream>>>(w, o, K);
  };
  pack(c1w1, sw_c1w1, FIN);
  pack(c1w2, sw_c1w2, HH);
  pack(c2aw1, sw_c2aw1, HH);  pack(c2aw2, sw_c2aw2, HH);
  pack(c2bw1, sw_c2bw1, HH);  pack(c2bw2, sw_c2bw2, HH);
  pack(c3aw1, sw_c3aw1, HH);  pack(c3aw2, sw_c3aw2, HH);
  pack(c3bw1, sw_c3bw1, HH);  pack(c3bw2, sw_c3bw2, HH);
  pack(ow1, sw_ow1, HH);

  // ---- 2) GIN conv helper ----
  auto conv = [&](const float* hin, int Kin, const unsigned* w1s, const float* b1,
                  const unsigned* w2s, const float* b2, float* hout, int relu) {
    size_t elems = (size_t)NN * Kin;
    copy4_kernel<<<(unsigned)((elems / 4 + 255) / 256), 256, 0, stream>>>(zb, hin, elems / 4);
    unsigned ethreads = (unsigned)EE * (Kin / 4);
    edge_agg_kernel<<<(ethreads + 255) / 256, 256, 0, stream>>>(hin, src, dstp, zb, Kin);
    gin_mlp_kernel<<<NN / 16, 256, 0, stream>>>(zb, Kin, w1s, b1, w2s, b2, hout, relu);
  };

  const size_t nh = (size_t)NN * HH;
  auto bnrelu = [&](float* h, const float* g, const float* b, const float* m,
                    const float* v) {
    bn_relu_kernel<<<(unsigned)((nh + 255) / 256), 256, 0, stream>>>(h, g, b, m, v, nh);
  };

  // ---- 3) forward pass ----
  conv(x, FIN, sw_c1w1, c1b1, sw_c1w2, c1b2, hA, 0);       // GIN block 1
  bnrelu(hA, bn1g, bn1b, bn1m, bn1v);
  conv(hA, HH, sw_c2aw1, c2ab1, sw_c2aw2, c2ab2, hB, 1);   // GIN block 2
  conv(hB, HH, sw_c2bw1, c2bb1, sw_c2bw2, c2bb2, hA, 0);
  bnrelu(hA, bn2g, bn2b, bn2m, bn2v);
  conv(hA, HH, sw_c3aw1, c3ab1, sw_c3aw2, c3ab2, hB, 1);   // GIN block 3
  conv(hB, HH, sw_c3bw1, c3bb1, sw_c3bw2, c3bb2, hA, 0);
  bnrelu(hA, bn3g, bn3b, bn3m, bn3v);

  // ---- 4) global mean pool ----
  zero_kernel<<<(GG * HH + 255) / 256, 256, 0, stream>>>(sums, (size_t)GG * HH);
  zero_kernel<<<(GG + 255) / 256, 256, 0, stream>>>(cnt, (size_t)GG);
  pool_acc_kernel<<<(NN * 32 + 255) / 256, 256, 0, stream>>>(hA, batch, sums, cnt);
  pool_div_kernel<<<(GG * HH + 255) / 256, 256, 0, stream>>>(sums, cnt, GG * HH);

  // ---- 5) output MLP ----
  gemm_relu_kernel<<<GG / 16, 256, 0, stream>>>(sums, sw_ow1, ob1, tpool);
  head_kernel<<<(GG + 255) / 256, 256, 0, stream>>>(tpool, ow2, ob2, (float*)d_out);
}
